// SparseTernaryViT_70961449664520
// MI455X (gfx1250) — compile-verified, matrix operand order checked by reading, not run on hardware
//
#include <hip/hip_runtime.h>
#include <hip/hip_bf16.h>

typedef __bf16 bf16;
typedef __attribute__((ext_vector_type(16))) bf16  v16bf;
typedef __attribute__((ext_vector_type(8)))  bf16  v8bf;
typedef __attribute__((ext_vector_type(8)))  float v8f;
typedef __attribute__((ext_vector_type(4)))  unsigned v4u;
typedef __attribute__((ext_vector_type(8)))  int   v8i;
typedef __attribute__((ext_vector_type(4)))  int   v4i;

#define DEPTH   12
#define HEADS   12
#define DHD     64
#define DIM     768
#define FFDIM   3072
#define NB      32
#define TOK     197
#define ROWS    (NB * TOK)        // 6304
#define NPATCH  (NB * 196)        // 6272
#define SSTRIDE 224               // padded attention row stride
#define NMAT    48

#define BM 128
#define BN 128
#define BK 32
#define KPAD 8
#define LDT (BK + KPAD)

// ---------------- helpers ----------------

__device__ __forceinline__ float gelu_tanh(float x) {
    float x3 = x * x * x;
    return 0.5f * x * (1.0f + tanhf(0.7978845608028654f * (x + 0.044715f * x3)));
}

__device__ __forceinline__ float block_sum256(float v) {
    __shared__ float red[256];
    int tid = threadIdx.x;
    red[tid] = v;
    __syncthreads();
    for (int s = 128; s > 0; s >>= 1) {
        if (tid < s) red[tid] += red[tid + s];
        __syncthreads();
    }
    float r = red[0];
    __syncthreads();
    return r;
}

__device__ __forceinline__ float block_max256(float v) {
    __shared__ float red[256];
    int tid = threadIdx.x;
    red[tid] = v;
    __syncthreads();
    for (int s = 128; s > 0; s >>= 1) {
        if (tid < s) red[tid] = fmaxf(red[tid], red[tid + s]);
        __syncthreads();
    }
    float r = red[0];
    __syncthreads();
    return r;
}

// ---------------- generic bf16 WMMA GEMM ----------------
// C[M,N] = alpha * A[M,K] x B[K,N] + bias(+resid), optional GELU,
// outputs to fp32 and/or bf16.  Batched via blockIdx.z with element strides.
// A tile is fetched by the Tensor Data Mover (tensor_load_to_lds) issued by
// wave 0, with D# LDS padding reproducing the +8 element row pad; B tile is
// loaded cooperatively and stored transposed.

__global__ __launch_bounds__(256, 2)
void gemm_bf16_kernel(const bf16* __restrict__ A, long long bsA, int lda,
                      const bf16* __restrict__ B, long long bsB, int ldb,
                      const float* __restrict__ bias,
                      const float* __restrict__ resid,
                      float* __restrict__ outF, bf16* __restrict__ outH,
                      long long bsC, int ldc,
                      int M, int N, int K, float alpha, int fuse_gelu)
{
    __shared__ __align__(16) bf16 sA[BM][LDT];
    __shared__ __align__(16) bf16 sB[BN][LDT];   // stored transposed: [n][k]

    const int tid  = threadIdx.x;
    const int lane = tid & 31;
    const int wave = tid >> 5;        // 0..7
    const int wm   = wave >> 1;       // 0..3  (M sub-tiles of 32)
    const int wn   = wave & 1;        // 0..1  (N sub-tiles of 64)
    const int bm0  = blockIdx.x * BM;
    const int bn0  = blockIdx.y * BN;
    const long long bz = blockIdx.z;

    A += bz * bsA;
    B += bz * bsB;
    if (outF)  outF  += bz * bsC;
    if (outH)  outH  += bz * bsC;
    if (resid) resid += bz * bsC;

    v8f acc[2][4] = {};

    const bool vecB = ((ldb & 7) == 0);

    // Uniform pieces of the TDM descriptor (D#).
    const unsigned ldsA =
        (unsigned)__builtin_amdgcn_readfirstlane(
            (int)(unsigned)(unsigned long long)(void*)&sA[0][0]);

    for (int k0 = 0; k0 < K; k0 += BK) {
        // ---- A tile (BM x BK) via Tensor Data Mover, wave 0 issues ----
        if (tid < 32) {
            unsigned long long ga =
                (unsigned long long)(const void*)(A + (long long)bm0 * lda + k0);
            unsigned ga_lo = (unsigned)__builtin_amdgcn_readfirstlane((int)(unsigned)ga);
            unsigned ga_hi = (unsigned)__builtin_amdgcn_readfirstlane((int)(unsigned)(ga >> 32));
            unsigned td0 = (unsigned)(K - k0);      // tensor_dim0 (elements)
            unsigned td1 = (unsigned)(M - bm0);     // tensor_dim1 (rows)
            unsigned long long st0 = (unsigned long long)(unsigned)lda;

            v4u g0;
            g0[0] = 1u;                                   // count=1, user mode
            g0[1] = ldsA;                                 // lds_addr (bytes)
            g0[2] = ga_lo;                                // global_addr[31:0]
            g0[3] = (ga_hi & 0x01FFFFFFu) | (2u << 30);   // addr[56:32], type=2

            v8i g1;
            // data_size=2B(code 1), pad_enable=1, pad_interval=16DW(code 3),
            // pad_amount=4DW(code 3)  -> reproduces LDT = BK+8 elements
            g1[0] = (int)((1u << 16) | (1u << 20) | (3u << 22) | (3u << 25));
            g1[1] = (int)((td0 & 0xFFFFu) << 16);                       // dim0 lo16
            g1[2] = (int)(((td0 >> 16) & 0xFFFFu) | ((td1 & 0xFFFFu) << 16));
            g1[3] = (int)(((td1 >> 16) & 0xFFFFu) | ((unsigned)BK << 16)); // tile_dim0
            g1[4] = (int)(unsigned)BM;                                  // tile_dim1
            g1[5] = (int)(unsigned)(st0 & 0xFFFFFFFFu);                 // stride lo32
            g1[6] = (int)(unsigned)((st0 >> 32) & 0xFFFFu);             // stride hi16
            g1[7] = 0;

            v4i gz4 = {0, 0, 0, 0};
#if defined(__clang_major__) && (__clang_major__ >= 23)
            v8i gz8 = {0, 0, 0, 0, 0, 0, 0, 0};
            __builtin_amdgcn_tensor_load_to_lds(g0, g1, gz4, gz4, gz8, 0);
#else
            __builtin_amdgcn_tensor_load_to_lds(g0, g1, gz4, gz4, 0);
#endif
        }

        // ---- B tile (BK x BN) cooperatively, transposed into sB[n][k] ----
        for (int i = tid; i < (BN * BK) / 8; i += 256) {
            int kk = i >> 4;                // BN/8 = 16 chunks per k-row
            int n0 = (i & 15) << 3;
            int gk = k0 + kk, gn = bn0 + n0;
            v8bf val;
            if (vecB && gk < K && gn + 7 < N) {
                val = *(const v8bf*)(B + (long long)gk * ldb + gn);
            } else {
#pragma unroll
                for (int j = 0; j < 8; j++)
                    val[j] = (gk < K && gn + j < N)
                                 ? B[(long long)gk * ldb + gn + j] : (bf16)0.0f;
            }
#pragma unroll
            for (int j = 0; j < 8; j++) sB[n0 + j][kk] = val[j];
        }

        // speculative prefetch of next B K-tile (global_prefetch_b8)
        if (k0 + BK + (tid & 31) < K)
            __builtin_prefetch(B + (long long)(k0 + BK + (tid & 31)) * ldb + bn0, 0, 1);

        // wave 0 drains its TDM op, then the barrier publishes the tiles
        if (tid < 32) __builtin_amdgcn_s_wait_tensorcnt(0);
        __syncthreads();

        // ---- fragments + WMMA ----
        const int lrow = lane & 15;
        const int lhi  = (lane >> 4) << 3;   // K chunk base: 0 or 8
        v16bf af[2], bfm[4];
#pragma unroll
        for (int mt = 0; mt < 2; mt++) {
            const bf16* p = &sA[wm * 32 + mt * 16 + lrow][lhi];
            v8bf lo = *(const v8bf*)p;
            v8bf hi = *(const v8bf*)(p + 16);
            af[mt] = __builtin_shufflevector(lo, hi, 0, 1, 2, 3, 4, 5, 6, 7,
                                             8, 9, 10, 11, 12, 13, 14, 15);
        }
#pragma unroll
        for (int nt = 0; nt < 4; nt++) {
            const bf16* p = &sB[wn * 64 + nt * 16 + lrow][lhi];
            v8bf lo = *(const v8bf*)p;
            v8bf hi = *(const v8bf*)(p + 16);
            bfm[nt] = __builtin_shufflevector(lo, hi, 0, 1, 2, 3, 4, 5, 6, 7,
                                              8, 9, 10, 11, 12, 13, 14, 15);
        }
#pragma unroll
        for (int mt = 0; mt < 2; mt++)
#pragma unroll
            for (int nt = 0; nt < 4; nt++)
                acc[mt][nt] = __builtin_amdgcn_wmma_f32_16x16x32_bf16(
                    false, af[mt], false, bfm[nt], (short)0, acc[mt][nt],
                    false, false);
        __syncthreads();
    }

    // ---- epilogue (C/D layout: VGPR i <-> rows i / i+8; lane&15 = col) ----
#pragma unroll
    for (int mt = 0; mt < 2; mt++) {
#pragma unroll
        for (int nt = 0; nt < 4; nt++) {
            int col = bn0 + wn * 64 + nt * 16 + (lane & 15);
            if (col >= N) continue;
            float bval = bias ? bias[col] : 0.0f;
#pragma unroll
            for (int i = 0; i < 8; i++) {
                int row = bm0 + wm * 32 + mt * 16 + ((lane >> 4) << 3) + i;
                if (row >= M) continue;
                float v = acc[mt][nt][i] * alpha + bval;
                if (fuse_gelu) v = gelu_tanh(v);
                long long o = (long long)row * ldc + col;
                if (resid) v += resid[o];
                if (outF) outF[o] = v;
                if (outH) outH[o] = (bf16)v;
            }
        }
    }
}

// ---------------- patch extraction + LayerNorm -> bf16 ----------------

__global__ __launch_bounds__(256)
void patch_ln_kernel(const float* __restrict__ img,
                     const float* __restrict__ g, const float* __restrict__ b,
                     bf16* __restrict__ out)
{
    int p = blockIdx.x;                 // 0..6271 : (batch, patch)
    int bi = p / 196, pi = p % 196;
    int ph = pi / 14, pw = pi % 14;
    float v[3];
    float s = 0.0f, s2 = 0.0f;
#pragma unroll
    for (int j = 0; j < 3; j++) {
        int e  = threadIdx.x + j * 256;
        int c  = e % 3;
        int p2 = (e / 3) % 16;
        int p1 = e / 48;
        float x = img[(((long long)bi * 3 + c) * 224 + (ph * 16 + p1)) * 224
                      + (pw * 16 + p2)];
        v[j] = x; s += x; s2 += x * x;
    }
    s  = block_sum256(s);
    s2 = block_sum256(s2);
    float mu   = s / 768.0f;
    float rstd = rsqrtf(s2 / 768.0f - mu * mu + 1e-5f);
#pragma unroll
    for (int j = 0; j < 3; j++) {
        int e = threadIdx.x + j * 256;
        out[(long long)p * DIM + e] = (bf16)((v[j] - mu) * rstd * g[e] + b[e]);
    }
}

// ---------------- embedding LN + cls + pos -> x fp32 ----------------

__global__ __launch_bounds__(256)
void embed_kernel(const float* __restrict__ proj,
                  const float* __restrict__ g, const float* __restrict__ b,
                  const float* __restrict__ cls, const float* __restrict__ pos,
                  float* __restrict__ x)
{
    int r = blockIdx.x;                 // 0..6303
    int bb = r / TOK, t = r % TOK;
    if (t == 0) {
#pragma unroll
        for (int j = 0; j < 3; j++) {
            int e = threadIdx.x + j * 256;
            x[(long long)r * DIM + e] = cls[e] + pos[e];
        }
        return;
    }
    const float* src = proj + (long long)(bb * 196 + (t - 1)) * DIM;
    float v[3];
    float s = 0.0f, s2 = 0.0f;
#pragma unroll
    for (int j = 0; j < 3; j++) {
        int e = threadIdx.x + j * 256;
        float xv = src[e];
        v[j] = xv; s += xv; s2 += xv * xv;
    }
    s  = block_sum256(s);
    s2 = block_sum256(s2);
    float mu   = s / 768.0f;
    float rstd = rsqrtf(s2 / 768.0f - mu * mu + 1e-5f);
#pragma unroll
    for (int j = 0; j < 3; j++) {
        int e = threadIdx.x + j * 256;
        x[(long long)r * DIM + e] =
            (v[j] - mu) * rstd * g[e] + b[e] + pos[(long long)t * DIM + e];
    }
}

// ---------------- 768-wide LayerNorm (templated output) ----------------

template <typename OUT>
__global__ __launch_bounds__(256)
void ln768_kernel(const float* __restrict__ in,
                  const float* __restrict__ g, const float* __restrict__ b,
                  OUT* __restrict__ out)
{
    int r = blockIdx.x;
    const float* src = in + (long long)r * DIM;
    float v[3];
    float s = 0.0f, s2 = 0.0f;
#pragma unroll
    for (int j = 0; j < 3; j++) {
        int e = threadIdx.x + j * 256;
        float xv = src[e];
        v[j] = xv; s += xv; s2 += xv * xv;
    }
    s  = block_sum256(s);
    s2 = block_sum256(s2);
    float mu   = s / 768.0f;
    float rstd = rsqrtf(s2 / 768.0f - mu * mu + 1e-5f);
#pragma unroll
    for (int j = 0; j < 3; j++) {
        int e = threadIdx.x + j * 256;
        out[(long long)r * DIM + e] = (OUT)((v[j] - mu) * rstd * g[e] + b[e]);
    }
}

// ---------------- qkv split / head merge ----------------

__global__ void split_qkv_kernel(const float* __restrict__ qkv,
                                 bf16* __restrict__ q, bf16* __restrict__ kt,
                                 bf16* __restrict__ v)
{
    long long idx = (long long)blockIdx.x * 256 + threadIdx.x;
    long long total = (long long)ROWS * (3 * DIM);
    if (idx >= total) return;
    int c = (int)(idx % (3 * DIM));
    long long bt = idx / (3 * DIM);
    int b = (int)(bt / TOK), t = (int)(bt % TOK);
    int which = c / DIM, h = (c % DIM) / DHD, d = c % DHD;
    bf16 val = (bf16)qkv[idx];
    long long bh = (long long)b * HEADS + h;
    if (which == 0)      q [(bh * TOK + t) * DHD + d] = val;
    else if (which == 1) kt[(bh * DHD + d) * TOK + t] = val;
    else                 v [(bh * TOK + t) * DHD + d] = val;
}

__global__ void merge_heads_kernel(const float* __restrict__ oh,
                                   bf16* __restrict__ out)
{
    long long idx = (long long)blockIdx.x * 256 + threadIdx.x;
    long long total = (long long)ROWS * DIM;
    if (idx >= total) return;
    int c = (int)(idx % DIM);
    long long bt = idx / DIM;
    int b = (int)(bt / TOK), t = (int)(bt % TOK);
    int h = c / DHD, d = c % DHD;
    out[idx] = (bf16)oh[(((long long)b * HEADS + h) * TOK + t) * DHD + d];
}

// ---------------- softmax over padded 197-wide rows ----------------

__global__ __launch_bounds__(256)
void softmax_kernel(const float* __restrict__ s, bf16* __restrict__ p)
{
    long long r = blockIdx.x;          // B*H*197 rows
    const float* row = s + r * SSTRIDE;
    int tid = threadIdx.x;
    float v = (tid < TOK) ? row[tid] : -3.0e38f;
    float m = block_max256(v);
    float e = (tid < TOK) ? __expf(v - m) : 0.0f;
    float sum = block_sum256(e);
    float inv = 1.0f / sum;
    if (tid < SSTRIDE)
        p[r * SSTRIDE + tid] = (bf16)((tid < TOK) ? e * inv : 0.0f);
}

// ---------------- ternary quantization ----------------

__global__ void zero_u32_kernel(unsigned* __restrict__ p, long long n)
{
    for (long long i = (long long)blockIdx.x * 256 + threadIdx.x; i < n;
         i += (long long)gridDim.x * 256)
        p[i] = 0u;
}

__global__ __launch_bounds__(256)
void quant_stats_kernel(const float* __restrict__ w, long long n,
                        float* __restrict__ sum, unsigned* __restrict__ mx)
{
    float s = 0.0f, m = 0.0f;
    for (long long i = (long long)blockIdx.x * 256 + threadIdx.x; i < n;
         i += (long long)gridDim.x * 256) {
        float a = fabsf(w[i]);
        s += a; m = fmaxf(m, a);
    }
    s = block_sum256(s);
    m = block_max256(m);
    if (threadIdx.x == 0) {
        atomicAdd(sum, s);
        atomicMax(mx, __float_as_uint(m));
    }
}

__global__ __launch_bounds__(256)
void quant_hist_kernel(const float* __restrict__ w, long long n,
                       const unsigned* __restrict__ mx,
                       unsigned* __restrict__ bins)
{
    float m = __uint_as_float(*mx);
    float inv = (m > 0.0f) ? (256.0f / m) : 0.0f;
    for (long long i = (long long)blockIdx.x * 256 + threadIdx.x; i < n;
         i += (long long)gridDim.x * 256) {
        int b = (int)(fabsf(w[i]) * inv);
        if (b > 255) b = 255;
        atomicAdd(&bins[b], 1u);
    }
}

__global__ void quant_thresh_kernel(const unsigned* __restrict__ bins,
                                    const float* __restrict__ sum,
                                    const unsigned* __restrict__ mx,
                                    long long n, float sparsity,
                                    float* __restrict__ thr_scale)
{
    unsigned long long target =
        (unsigned long long)(sparsity * (double)n);
    unsigned long long cum = 0;
    int b = 0;
    for (; b < 256; b++) {
        cum += bins[b];
        if (cum >= target) break;
    }
    float m = __uint_as_float(*mx);
    thr_scale[0] = ((float)b + 0.5f) * m * (1.0f / 256.0f);  // ~quantile
    thr_scale[1] = sum[0] / (float)n;                        // mean(|W|)
}

__global__ void ternarize_kernel(const float* __restrict__ w,
                                 bf16* __restrict__ o, long long n,
                                 const float* __restrict__ ts)
{
    float thr = ts[0], sc = ts[1];
    for (long long i = (long long)blockIdx.x * 256 + threadIdx.x; i < n;
         i += (long long)gridDim.x * 256) {
        float x = w[i];
        float q = (fabsf(x) >= thr) ? (x > 0.0f ? sc : -sc) : 0.0f;
        o[i] = (bf16)q;
    }
}

__global__ void to_bf16_kernel(const float* __restrict__ w,
                               bf16* __restrict__ o, long long n)
{
    for (long long i = (long long)blockIdx.x * 256 + threadIdx.x; i < n;
         i += (long long)gridDim.x * 256)
        o[i] = (bf16)w[i];
}

// ---------------- host orchestration ----------------

extern "C" void kernel_launch(void* const* d_in, const int* in_sizes, int n_in,
                              void* d_out, int out_size, void* d_ws,
                              size_t ws_size, hipStream_t stream)
{
    (void)in_sizes; (void)n_in; (void)out_size; (void)ws_size;

    const float* img        = (const float*)d_in[0];
    const float* patch_ln_g = (const float*)d_in[1];
    const float* patch_ln_b = (const float*)d_in[2];
    const float* W_patch    = (const float*)d_in[3];
    const float* b_patch    = (const float*)d_in[4];
    const float* emb_ln_g   = (const float*)d_in[5];
    const float* emb_ln_b   = (const float*)d_in[6];
    const float* pos_emb    = (const float*)d_in[7];
    const float* cls_token  = (const float*)d_in[8];
    const float* ln1_g      = (const float*)d_in[9];
    const float* ln1_b      = (const float*)d_in[10];
    const float* Wqkv       = (const float*)d_in[11];
    const float* b_qkv      = (const float*)d_in[12];
    const float* Wo         = (const float*)d_in[13];
    const float* b_o        = (const float*)d_in[14];
    const float* ln2_g      = (const float*)d_in[15];
    const float* ln2_b      = (const float*)d_in[16];
    const float* W1         = (const float*)d_in[17];
    const float* b1         = (const float*)d_in[18];
    const float* W2         = (const float*)d_in[19];
    const float* b2         = (const float*)d_in[20];
    const float* norm_g     = (const float*)d_in[21];
    const float* norm_b     = (const float*)d_in[22];

    float* out = (float*)d_out;

    // ---- workspace carve-up ----
    char* base = (char*)d_ws;
    size_t off = 0;
    auto alloc = [&](size_t bytes) -> char* {
        char* p = base + off;
        off = (off + bytes + 255) & ~(size_t)255;
        return p;
    };
    float* x       = (float*)alloc((size_t)ROWS * DIM * 4);
    float* tmp     = (float*)alloc((size_t)ROWS * 3 * DIM * 4);
    bf16*  hbuf    = (bf16*) alloc((size_t)ROWS * DIM * 2);
    bf16*  ffh     = (bf16*) alloc((size_t)ROWS * FFDIM * 2);
    bf16*  qb      = (bf16*) alloc((size_t)NB * HEADS * TOK * DHD * 2);
    bf16*  ktb     = (bf16*) alloc((size_t)NB * HEADS * DHD * TOK * 2);
    bf16*  vb      = (bf16*) alloc((size_t)NB * HEADS * TOK * DHD * 2);
    float* scores  = (float*)alloc((size_t)NB * HEADS * TOK * SSTRIDE * 4);
    bf16*  probs   = (bf16*) alloc((size_t)NB * HEADS * TOK * SSTRIDE * 2);
    float* oheads  = (float*)alloc((size_t)NB * HEADS * TOK * DHD * 4);
    bf16*  wq_qkv  = (bf16*) alloc((size_t)DEPTH * DIM * 3 * DIM * 2);
    bf16*  wq_o    = (bf16*) alloc((size_t)DEPTH * DIM * DIM * 2);
    bf16*  wq_1    = (bf16*) alloc((size_t)DEPTH * DIM * FFDIM * 2);
    bf16*  wq_2    = (bf16*) alloc((size_t)DEPTH * FFDIM * DIM * 2);
    bf16*  wpatchb = (bf16*) alloc((size_t)DIM * DIM * 2);
    float* stats   = (float*)alloc((size_t)NMAT * 4 * 4 + (size_t)NMAT * 256 * 4);
    unsigned* bins = (unsigned*)(stats + NMAT * 4);

    // ---- GEMM launcher ----
    auto gemm = [&](const bf16* A, long long bsA, int lda,
                    const bf16* B, long long bsB, int ldb,
                    const float* bias, const float* resid,
                    float* outF, bf16* outH, long long bsC, int ldc,
                    int M, int N, int K, float alpha, int gelu, int batch) {
        dim3 grid((M + BM - 1) / BM, (N + BN - 1) / BN, batch);
        gemm_bf16_kernel<<<grid, dim3(256), 0, stream>>>(
            A, bsA, lda, B, bsB, ldb, bias, resid, outF, outH, bsC, ldc,
            M, N, K, alpha, gelu);
    };

    // ---- ternary weight quantization ----
    zero_u32_kernel<<<64, 256, 0, stream>>>(
        (unsigned*)stats, (long long)(NMAT * 4 + NMAT * 256));

    auto quantize = [&](const float* W, long long n, int slot, bf16* dst) {
        float*    st = stats + slot * 4;
        unsigned* bn = bins + slot * 256;
        quant_stats_kernel<<<256, 256, 0, stream>>>(W, n, st, (unsigned*)(st + 1));
        quant_hist_kernel <<<256, 256, 0, stream>>>(W, n, (const unsigned*)(st + 1), bn);
        quant_thresh_kernel<<<1, 1, 0, stream>>>(bn, st, (const unsigned*)(st + 1),
                                                 n, 0.5f, st + 2);
        ternarize_kernel<<<1024, 256, 0, stream>>>(W, dst, n, st + 2);
    };

    const long long nQKV = (long long)DIM * 3 * DIM;
    const long long nO   = (long long)DIM * DIM;
    const long long nF   = (long long)DIM * FFDIM;

    for (int i = 0; i < DEPTH; i++) {
        quantize(Wqkv + (long long)i * nQKV, nQKV, i,      wq_qkv + (long long)i * nQKV);
        quantize(Wo   + (long long)i * nO,   nO,   12 + i, wq_o   + (long long)i * nO);
        if (i < DEPTH - 1) {
            quantize(W1 + (long long)i * nF, nF, 24 + i, wq_1 + (long long)i * nF);
            quantize(W2 + (long long)i * nF, nF, 36 + i, wq_2 + (long long)i * nF);
        } else {
            to_bf16_kernel<<<1024, 256, 0, stream>>>(W1 + (long long)i * nF,
                                                     wq_1 + (long long)i * nF, nF);
            to_bf16_kernel<<<1024, 256, 0, stream>>>(W2 + (long long)i * nF,
                                                     wq_2 + (long long)i * nF, nF);
        }
    }
    to_bf16_kernel<<<1024, 256, 0, stream>>>(W_patch, wpatchb, nO);

    // ---- patch embedding ----
    patch_ln_kernel<<<NPATCH, 256, 0, stream>>>(img, patch_ln_g, patch_ln_b, hbuf);
    gemm(hbuf, 0, DIM, wpatchb, 0, DIM, b_patch, nullptr, tmp, nullptr, 0, DIM,
         NPATCH, DIM, DIM, 1.0f, 0, 1);
    embed_kernel<<<ROWS, 256, 0, stream>>>(tmp, emb_ln_g, emb_ln_b,
                                           cls_token, pos_emb, x);

    // ---- transformer layers ----
    const int BHT = NB * HEADS;                 // 384 attention batches
    const long long sQ = (long long)TOK * DHD;  // q / v / oheads batch stride
    const long long sK = (long long)DHD * TOK;  // kt batch stride
    const long long sS = (long long)TOK * SSTRIDE;

    for (int i = 0; i < DEPTH; i++) {
        // attention
        ln768_kernel<bf16><<<ROWS, 256, 0, stream>>>(
            x, ln1_g + i * DIM, ln1_b + i * DIM, hbuf);
        gemm(hbuf, 0, DIM, wq_qkv + (long long)i * nQKV, 0, 3 * DIM,
             b_qkv + (long long)i * 3 * DIM, nullptr, tmp, nullptr, 0, 3 * DIM,
             ROWS, 3 * DIM, DIM, 1.0f, 0, 1);
        {
            long long tot = (long long)ROWS * 3 * DIM;
            split_qkv_kernel<<<(unsigned)((tot + 255) / 256), 256, 0, stream>>>(
                tmp, qb, ktb, vb);
        }
        gemm(qb, sQ, DHD, ktb, sK, TOK, nullptr, nullptr, scores, nullptr,
             sS, SSTRIDE, TOK, TOK, DHD, 0.125f, 0, BHT);
        softmax_kernel<<<BHT * TOK, 256, 0, stream>>>(scores, probs);
        gemm(probs, sS, SSTRIDE, vb, sQ, DHD, nullptr, nullptr, oheads, nullptr,
             sQ, DHD, TOK, DHD, TOK, 1.0f, 0, BHT);
        {
            long long tot = (long long)ROWS * DIM;
            merge_heads_kernel<<<(unsigned)((tot + 255) / 256), 256, 0, stream>>>(
                oheads, hbuf);
        }
        gemm(hbuf, 0, DIM, wq_o + (long long)i * nO, 0, DIM,
             b_o + (long long)i * DIM, x, x, nullptr, 0, DIM,
             ROWS, DIM, DIM, 1.0f, 0, 1);

        // MLP
        ln768_kernel<bf16><<<ROWS, 256, 0, stream>>>(
            x, ln2_g + i * DIM, ln2_b + i * DIM, hbuf);
        gemm(hbuf, 0, DIM, wq_1 + (long long)i * nF, 0, FFDIM,
             b1 + (long long)i * FFDIM, nullptr, nullptr, ffh, 0, FFDIM,
             ROWS, FFDIM, DIM, 1.0f, 1, 1);
        gemm(ffh, 0, FFDIM, wq_2 + (long long)i * nF, 0, DIM,
             b2 + (long long)i * DIM, x, x, nullptr, 0, DIM,
             ROWS, DIM, FFDIM, 1.0f, 0, 1);
    }

    // ---- final LayerNorm -> output (fp32) ----
    ln768_kernel<float><<<ROWS, 256, 0, stream>>>(x, norm_g, norm_b, out);
}
